// VAE_69990787056315
// MI455X (gfx1250) — matrix operand order repaired and checked by reading, not compiled, not run on hardware
//
#include <hip/hip_runtime.h>
#include <hip/hip_bf16.h>
#include <math.h>

typedef __bf16 bf16_t;
typedef __attribute__((ext_vector_type(16))) __bf16 v16bf;
typedef __attribute__((ext_vector_type(8)))  __bf16 v8bf;
typedef __attribute__((ext_vector_type(8)))  float  v8f;

// Global (address_space(1)) pointer types so GEMM loads lower to
// global_load_b128 (LOADcnt only) instead of flat_load_b128 (LOADcnt+DScnt).
typedef const __attribute__((address_space(1))) bf16_t* gbf_ptr;
typedef const __attribute__((address_space(1))) v8bf*   gv8_ptr;
typedef const __attribute__((address_space(1))) v16bf*  gv16_ptr;

#define BATCH 256
#define GHN 5
#define GWN 7
#define NTOK 35
#define DIM 512
#define NHEAD 8
#define HDIM 64
#define ROTD 32
#define KCB 84
#define KCBP 96
#define FFD 2048
#define NROWS (BATCH*NTOK)      /* 8960 */
#define PPK 1200
#define PPKP 1216

// ---------------------------------------------------------------------------
// WMMA fragment builders (bf16, per CDNA5 16x16x32 layouts)
// A: 16x32 (MxK) row-major activation, lda = K
// B: 32x16 (KxN) taken from transposed weight storage WT[N][K]
// Per-lane base pointers are hoisted; in-loop loads use immediate offsets.
// ---------------------------------------------------------------------------
__device__ __forceinline__ v16bf make_a(gbf_ptr p) {
  v8bf lo = *(gv8_ptr)p;          // K = kq .. kq+7
  v8bf hi = *(gv8_ptr)(p + 16);   // K = 16+kq .. 23+kq
  v16bf out;
#pragma unroll
  for (int i = 0; i < 8; ++i) { out[i] = lo[i]; out[i + 8] = hi[i]; }
  return out;
}

__device__ __forceinline__ void wmma_all(const v16bf a[4], const v16bf b[4],
                                         v8f acc[4][4]) {
#pragma unroll
  for (int i = 0; i < 4; ++i)
#pragma unroll
    for (int j = 0; j < 4; ++j)
      acc[i][j] = __builtin_amdgcn_wmma_f32_16x16x32_bf16(
          false, a[i], false, b[j], (short)0, acc[i][j], false, false);
}

// Epilogues: 0=bias->f32  1=bias+gelu->bf16  2=bias+residual->f32
//            3=bias+pos->f32  4=plain->f32   5=bias+tanh pixel scatter
// FULL: N % 128 == 0 (no column predication at all)
template <int EPI, bool FULL>
__global__ __launch_bounds__(128) void gemm_bf16(
    const bf16_t* __restrict__ A, const bf16_t* __restrict__ BT,
    float* C, bf16_t* Cb, const float* __restrict__ bias,
    const float* extra, int N, int K, int ldc) {
  int lane = threadIdx.x & 31;
  int wid  = threadIdx.x >> 5;
  int row0 = blockIdx.y * 128 + (wid >> 1) * 64;
  int col0 = blockIdx.x * 128 + (wid & 1) * 64;

  // Column-tile validity as wave-uniform scalars; clamp loads of invalid
  // tiles to column 0 (valid memory, results discarded at store time).
  int vn[4];
  int colL[4];
#pragma unroll
  for (int j = 0; j < 4; ++j) {
    if (FULL) {
      vn[j] = 1;
      colL[j] = col0 + j * 16;
    } else {
      vn[j] = __builtin_amdgcn_readfirstlane((col0 + j * 16) < N ? 1 : 0);
      colL[j] = vn[j] ? (col0 + j * 16) : 0;
    }
  }

  // Hoisted per-lane fragment base pointers (global address space).
  int ar = lane & 15;
  int kq = (lane >> 4) << 3;           // A: 0 or 8
  int kh = (lane >> 4) << 3;           // B: 0 or 8
  gbf_ptr pA[4];
  gbf_ptr pB[4];
#pragma unroll
  for (int i = 0; i < 4; ++i)
    pA[i] = (gbf_ptr)(A + (size_t)(row0 + i * 16 + ar) * K + kq);
#pragma unroll
  for (int j = 0; j < 4; ++j)
    pB[j] = (gbf_ptr)(BT + (size_t)(colL[j] + ar) * K + kh);

  v8f acc[4][4];
#pragma unroll
  for (int i = 0; i < 4; ++i)
#pragma unroll
    for (int j = 0; j < 4; ++j)
#pragma unroll
      for (int e = 0; e < 8; ++e) acc[i][j][e] = 0.f;

  // Register-double-buffered main loop, K unrolled by 64 (K/32 is even for
  // all our GEMMs: 512, 1216, 2048). In-loop loads use immediate offsets;
  // the only per-iteration address math is 8 pointer increments by +128B.
  // The final prefetch over-reads 64B past the row end (stays inside d_ws,
  // never stored).
  v16bf a0[4], b0[4], a1[4], b1[4];
#pragma unroll
  for (int i = 0; i < 4; ++i) a0[i] = make_a(pA[i]);
#pragma unroll
  for (int j = 0; j < 4; ++j) b0[j] = make_a(pB[j]);

  for (int k0 = 0; k0 < K; k0 += 64) {
#pragma unroll
    for (int i = 0; i < 4; ++i) a1[i] = make_a(pA[i] + 32);
#pragma unroll
    for (int j = 0; j < 4; ++j) b1[j] = make_a(pB[j] + 32);
    wmma_all(a0, b0, acc);
#pragma unroll
    for (int i = 0; i < 4; ++i) a0[i] = make_a(pA[i] + 64);
#pragma unroll
    for (int j = 0; j < 4; ++j) b0[j] = make_a(pB[j] + 64);
    wmma_all(a1, b1, acc);
#pragma unroll
    for (int i = 0; i < 4; ++i) pA[i] += 64;
#pragma unroll
    for (int j = 0; j < 4; ++j) pB[j] += 64;
  }

  int cn = lane & 15;
  int mb = (lane >> 4) * 8;
#pragma unroll
  for (int i = 0; i < 4; ++i) {
#pragma unroll
    for (int j = 0; j < 4; ++j) {
      if (!FULL && !vn[j]) continue;   // scalar (readfirstlane) branch
      int col = col0 + j * 16 + cn;
#pragma unroll
      for (int e = 0; e < 8; ++e) {
        int row   = row0 + i * 16 + mb + e;
        float val = acc[i][j][e];
        if (EPI == 0) {
          C[(size_t)row * ldc + col] = val + bias[col];
        } else if (EPI == 1) {
          float t = val + bias[col];
          float g = 0.5f * t * (1.f + erff(t * 0.70710678f));
          Cb[(size_t)row * ldc + col] = (bf16_t)g;
        } else if (EPI == 2) {
          C[(size_t)row * ldc + col] = extra[(size_t)row * ldc + col] + val + bias[col];
        } else if (EPI == 3) {
          C[(size_t)row * ldc + col] =
              val + bias[col] + extra[(size_t)(row % NTOK) * ldc + col];
        } else if (EPI == 4) {
          C[(size_t)row * ldc + col] = val;
        } else if (EPI == 5) {
          float t = tanhf(val + bias[col]);
          int bb = row / NTOK, tt = row % NTOK, gh = tt / GWN, gw = tt % GWN;
          int cc = col / 400, r2 = col % 400, ph = r2 / 20, pw = r2 % 20;
          C[(size_t)(((bb * 3 + cc) * 100) + gh * 20 + ph) * 140 + gw * 20 + pw] = t;
        }
      }
    }
  }
}

// ---------------------------------------------------------------------------
// LayerNorm (one row per block, 256 threads, D=512)
// ---------------------------------------------------------------------------
template <int OUTBF>
__global__ __launch_bounds__(256) void ln_kernel(const float* __restrict__ X,
                                                 const float* __restrict__ g,
                                                 const float* __restrict__ b,
                                                 float* of, bf16_t* ob) {
  __shared__ float red[256];
  int row = blockIdx.x, tid = threadIdx.x;
  const float* xr = X + (size_t)row * DIM;
  float v0 = xr[tid], v1 = xr[tid + 256];
  red[tid] = v0 + v1;
  __syncthreads();
  for (int s = 128; s > 0; s >>= 1) { if (tid < s) red[tid] += red[tid + s]; __syncthreads(); }
  float mu = red[0] * (1.f / 512.f);
  __syncthreads();
  float d0 = v0 - mu, d1 = v1 - mu;
  red[tid] = d0 * d0 + d1 * d1;
  __syncthreads();
  for (int s = 128; s > 0; s >>= 1) { if (tid < s) red[tid] += red[tid + s]; __syncthreads(); }
  float rstd = rsqrtf(red[0] * (1.f / 512.f) + 1e-6f);
  float o0 = d0 * rstd * g[tid] + b[tid];
  float o1 = d1 * rstd * g[tid + 256] + b[tid + 256];
  if (OUTBF == 0) {
    of[(size_t)row * DIM + tid] = o0;
    of[(size_t)row * DIM + tid + 256] = o1;
  } else {
    ob[(size_t)row * DIM + tid] = (bf16_t)o0;
    ob[(size_t)row * DIM + tid + 256] = (bf16_t)o1;
  }
}

// L2-normalize rows of z -> bf16
__global__ __launch_bounds__(256) void zn_kernel(const float* __restrict__ Z,
                                                 bf16_t* __restrict__ out) {
  __shared__ float red[256];
  int row = blockIdx.x, tid = threadIdx.x;
  const float* zr = Z + (size_t)row * DIM;
  float v0 = zr[tid], v1 = zr[tid + 256];
  red[tid] = v0 * v0 + v1 * v1;
  __syncthreads();
  for (int s = 128; s > 0; s >>= 1) { if (tid < s) red[tid] += red[tid + s]; __syncthreads(); }
  float inv = 1.f / fmaxf(sqrtf(red[0]), 1e-12f);
  out[(size_t)row * DIM + tid] = (bf16_t)(v0 * inv);
  out[(size_t)row * DIM + tid + 256] = (bf16_t)(v1 * inv);
}

// Normalize codebook rows -> en (f32) and enbf (bf16, padded to 96 rows)
__global__ __launch_bounds__(256) void en_kernel(const float* __restrict__ E,
                                                 float* __restrict__ en,
                                                 bf16_t* __restrict__ enbf) {
  __shared__ float red[256];
  int row = blockIdx.x, tid = threadIdx.x;
  if (row >= KCB) {  // zero pad rows 84..95 of enbf
    enbf[(size_t)row * DIM + tid] = (bf16_t)0.f;
    enbf[(size_t)row * DIM + tid + 256] = (bf16_t)0.f;
    return;
  }
  const float* er = E + (size_t)row * DIM;
  float v0 = er[tid], v1 = er[tid + 256];
  red[tid] = v0 * v0 + v1 * v1;
  __syncthreads();
  for (int s = 128; s > 0; s >>= 1) { if (tid < s) red[tid] += red[tid + s]; __syncthreads(); }
  float inv = 1.f / fmaxf(sqrtf(red[0]), 1e-12f);
  float o0 = v0 * inv, o1 = v1 * inv;
  en[(size_t)row * DIM + tid] = o0;
  en[(size_t)row * DIM + tid + 256] = o1;
  enbf[(size_t)row * DIM + tid] = (bf16_t)o0;
  enbf[(size_t)row * DIM + tid + 256] = (bf16_t)o1;
}

// ---------------------------------------------------------------------------
// RoPE + head split: qkv[8960,1536] -> qh/kh/vh [B,H,N,HD]
// ---------------------------------------------------------------------------
__global__ void rope_kernel(const float* __restrict__ qkv, float* __restrict__ qh,
                            float* __restrict__ kh, float* __restrict__ vh) {
  int gid = blockIdx.x * blockDim.x + threadIdx.x;
  if (gid >= BATCH * NHEAD * NTOK * HDIM) return;
  int hd = gid & 63;
  int n  = (gid >> 6) % NTOK;
  int h  = (gid / (64 * NTOK)) & 7;
  int bb = gid / (64 * NTOK * NHEAD);
  int row = bb * NTOK + n;
  const float* qr = qkv + (size_t)row * 1536;
  float q = qr[h * 64 + hd];
  float k = qr[512 + h * 64 + hd];
  float v = qr[1024 + h * 64 + hd];
  if (hd < ROTD) {
    int gh = n / GWN, gw = n % GWN;
    int j = ((hd < 16) ? hd : hd - 16) >> 1;
    float base = 3.14159265358979f * (1.f + (float)j * (16.5f / 7.f));
    float t = (hd < 16) ? (-1.f + 2.f * gh / (float)(GHN - 1))
                        : (-1.f + 2.f * gw / (float)(GWN - 1));
    float f = t * base;
    float c = cosf(f), s = sinf(f);
    float qp = qr[h * 64 + (hd ^ 1)];
    float kp = qr[512 + h * 64 + (hd ^ 1)];
    q = (hd & 1) ? (q * c + qp * s) : (q * c - qp * s);
    k = (hd & 1) ? (k * c + kp * s) : (k * c - kp * s);
  }
  qh[gid] = q; kh[gid] = k; vh[gid] = v;
}

// ---------------------------------------------------------------------------
// Attention: one block per (b,h); N=35, HD=64; output bf16 [8960,512]
// ---------------------------------------------------------------------------
__global__ __launch_bounds__(256) void attn_kernel(const float* __restrict__ qh,
                                                   const float* __restrict__ kh,
                                                   const float* __restrict__ vh,
                                                   bf16_t* __restrict__ O) {
  __shared__ float sq[NTOK * HDIM], sk[NTOK * HDIM], sv[NTOK * HDIM];
  __shared__ float sc[NTOK * NTOK];
  int bh = blockIdx.x, tid = threadIdx.x;
  int bb = bh >> 3, h = bh & 7;
  size_t base = (size_t)bh * NTOK * HDIM;
  for (int i = tid; i < NTOK * HDIM; i += 256) {
    sq[i] = qh[base + i]; sk[i] = kh[base + i]; sv[i] = vh[base + i];
  }
  __syncthreads();
  for (int e = tid; e < NTOK * NTOK; e += 256) {
    int i = e / NTOK, j = e % NTOK;
    float acc = 0.f;
#pragma unroll 8
    for (int d = 0; d < HDIM; ++d) acc += sq[i * HDIM + d] * sk[j * HDIM + d];
    sc[e] = acc * 0.125f;  // 1/sqrt(64)
  }
  __syncthreads();
  if (tid < NTOK) {
    float m = -1e30f;
    for (int j = 0; j < NTOK; ++j) m = fmaxf(m, sc[tid * NTOK + j]);
    float s = 0.f;
    for (int j = 0; j < NTOK; ++j) { float e = expf(sc[tid * NTOK + j] - m); sc[tid * NTOK + j] = e; s += e; }
    float inv = 1.f / s;
    for (int j = 0; j < NTOK; ++j) sc[tid * NTOK + j] *= inv;
  }
  __syncthreads();
  for (int e = tid; e < NTOK * HDIM; e += 256) {
    int n = e >> 6, hd = e & 63;
    float acc = 0.f;
    for (int m = 0; m < NTOK; ++m) acc += sc[n * NTOK + m] * sv[m * HDIM + hd];
    O[(size_t)(bb * NTOK + n) * DIM + h * 64 + hd] = (bf16_t)acc;
  }
}

// ---------------------------------------------------------------------------
// VQ: argmax over sim, idx out, x = en[idx]+pos, commit/count stats
// ---------------------------------------------------------------------------
__global__ __launch_bounds__(128) void vq_kernel(const float* __restrict__ sim,
                                                 const float* __restrict__ en,
                                                 const float* __restrict__ z,
                                                 const float* __restrict__ pos,
                                                 float* __restrict__ x,
                                                 int* __restrict__ idx_out,
                                                 float* commit, int* counts) {
  __shared__ float svv[128];
  __shared__ int sii[128];
  __shared__ int sidx;
  int row = blockIdx.x, tid = threadIdx.x;
  float v = (tid < KCB) ? sim[(size_t)row * KCBP + tid] : -1e30f;
  svv[tid] = v; sii[tid] = tid;
  __syncthreads();
  for (int s = 64; s > 0; s >>= 1) {
    if (tid < s) {
      if (svv[tid + s] > svv[tid] ||
          (svv[tid + s] == svv[tid] && sii[tid + s] < sii[tid])) {
        svv[tid] = svv[tid + s]; sii[tid] = sii[tid + s];
      }
    }
    __syncthreads();
  }
  if (tid == 0) { sidx = sii[0]; idx_out[row] = sidx; atomicAdd(&counts[sidx], 1); }
  __syncthreads();
  int idx = sidx;
  float local = 0.f;
  for (int j = tid; j < DIM; j += 128) {
    float q  = en[(size_t)idx * DIM + j];
    float zz = z[(size_t)row * DIM + j];
    x[(size_t)row * DIM + j] = q + pos[(size_t)(row % NTOK) * DIM + j];
    float d = q - zz;
    local += d * d;
  }
  svv[tid] = local;
  __syncthreads();
  for (int s = 64; s > 0; s >>= 1) { if (tid < s) svv[tid] += svv[tid + s]; __syncthreads(); }
  if (tid == 0) atomicAdd(commit, svv[0]);
}

__global__ void init_stats(float* commit, int* counts) {
  int t = threadIdx.x;
  if (t == 0) commit[0] = 0.f;
  if (t < KCB) counts[t] = 0;
}

__global__ void finalize_kernel(const float* commit, const int* counts, float* loss) {
  if (threadIdx.x == 0 && blockIdx.x == 0) {
    float total = 0.f;
    for (int k = 0; k < KCB; ++k) total += (float)counts[k];
    float denom = fmaxf(total, 1e-12f);
    float ent = 0.f;
    for (int k = 0; k < KCB; ++k) {
      float p = (float)counts[k] / denom;
      ent -= p * logf(fmaxf(p, 1e-12f));
    }
    float c = 0.25f * commit[0] / (float)((size_t)NROWS * DIM);
    loss[0] = c + 0.1f * (logf((float)KCB) - ent);
  }
}

// ---------------------------------------------------------------------------
// Weight transpose+convert: W[K,N] f32 -> WT[N,Kpad] bf16 (zero pad)
// ---------------------------------------------------------------------------
__global__ void convert_wT(const float* __restrict__ W, bf16_t* __restrict__ WT,
                           int K, int N, int Kpad) {
  int idx = blockIdx.x * blockDim.x + threadIdx.x;
  if (idx >= N * Kpad) return;
  int n = idx / Kpad, k = idx % Kpad;
  WT[idx] = (k < K) ? (bf16_t)W[(size_t)k * N + n] : (bf16_t)0.f;
}

// Patchify + normalize -> bf16 [8960,1216] (zero pad cols 1200..1215)
__global__ void patch_kernel(const float* __restrict__ x, bf16_t* __restrict__ P) {
  int idx = blockIdx.x * blockDim.x + threadIdx.x;
  if (idx >= NROWS * PPKP) return;
  int row = idx / PPKP, col = idx % PPKP;
  if (col >= PPK) { P[idx] = (bf16_t)0.f; return; }
  int bb = row / NTOK, t = row % NTOK, gh = t / GWN, gw = t % GWN;
  int c = col / 400, r2 = col % 400, ph = r2 / 20, pw = r2 % 20;
  float v = x[(size_t)(((bb * 3 + c) * 100) + gh * 20 + ph) * 140 + gw * 20 + pw];
  P[idx] = (bf16_t)(2.f * (v / 255.f - 0.5f));
}

// ---------------------------------------------------------------------------
extern "C" void kernel_launch(void* const* d_in, const int* in_sizes, int n_in,
                              void* d_out, int out_size, void* d_ws, size_t ws_size,
                              hipStream_t stream) {
  const float* x_in    = (const float*)d_in[0];
  const float* patch_w = (const float*)d_in[1];
  const float* patch_b = (const float*)d_in[2];
  const float* pos     = (const float*)d_in[3];
  const float* enc_g   = (const float*)d_in[16];
  const float* enc_b   = (const float*)d_in[17];
  const float* dec_g   = (const float*)d_in[30];
  const float* dec_b   = (const float*)d_in[31];
  const float* embed   = (const float*)d_in[32];
  const float* pred_w  = (const float*)d_in[33];
  const float* pred_b  = (const float*)d_in[34];

  struct BlkP { const float *ln1g,*ln1b,*qkvw,*qkvb,*projw,*projb,*ln2g,*ln2b,*w1,*b1,*w2,*b2; };
  auto getBlk = [&](int base, int blk) -> BlkP {
    BlkP p;
    p.ln1g  = (const float*)d_in[base + 0] + (size_t)blk * 512;
    p.ln1b  = (const float*)d_in[base + 1] + (size_t)blk * 512;
    p.qkvw  = (const float*)d_in[base + 2] + (size_t)blk * 512 * 1536;
    p.qkvb  = (const float*)d_in[base + 3] + (size_t)blk * 1536;
    p.projw = (const float*)d_in[base + 4] + (size_t)blk * 512 * 512;
    p.projb = (const float*)d_in[base + 5] + (size_t)blk * 512;
    p.ln2g  = (const float*)d_in[base + 6] + (size_t)blk * 512;
    p.ln2b  = (const float*)d_in[base + 7] + (size_t)blk * 512;
    p.w1    = (const float*)d_in[base + 8] + (size_t)blk * 512 * 2048;
    p.b1    = (const float*)d_in[base + 9] + (size_t)blk * 2048;
    p.w2    = (const float*)d_in[base + 10] + (size_t)blk * 2048 * 512;
    p.b2    = (const float*)d_in[base + 11] + (size_t)blk * 512;
    return p;
  };

  // ---- workspace layout ----
  char* wp = (char*)d_ws;
  auto alloc = [&](size_t bytes) -> char* {
    char* r = wp; wp += (bytes + 255) & ~(size_t)255; return r;
  };
  bf16_t* patchT = (bf16_t*)alloc((size_t)512 * PPKP * 2);
  bf16_t* predT  = (bf16_t*)alloc((size_t)1200 * 512 * 2);
  bf16_t* enbf   = (bf16_t*)alloc((size_t)KCBP * 512 * 2);
  bf16_t *qkvT[12], *projT[12], *w1T[12], *w2T[12];
  for (int i = 0; i < 12; ++i) {
    qkvT[i]  = (bf16_t*)alloc((size_t)1536 * 512 * 2);
    projT[i] = (bf16_t*)alloc((size_t)512 * 512 * 2);
    w1T[i]   = (bf16_t*)alloc((size_t)2048 * 512 * 2);
    w2T[i]   = (bf16_t*)alloc((size_t)512 * 2048 * 2);
  }
  bf16_t* patches = (bf16_t*)alloc((size_t)NROWS * PPKP * 2);
  float*  xst     = (float*)alloc((size_t)NROWS * DIM * 4);
  float*  zbuf    = (float*)alloc((size_t)NROWS * DIM * 4);
  bf16_t* hbuf    = (bf16_t*)alloc((size_t)NROWS * DIM * 2);
  float*  qkvbuf  = (float*)alloc((size_t)NROWS * 1536 * 4);  // aliased with ff1
  bf16_t* ff1buf  = (bf16_t*)qkvbuf;
  float*  qhb     = (float*)alloc((size_t)NROWS * DIM * 4);
  float*  khb     = (float*)alloc((size_t)NROWS * DIM * 4);
  float*  vhb     = (float*)alloc((size_t)NROWS * DIM * 4);
  bf16_t* Obuf    = (bf16_t*)alloc((size_t)NROWS * DIM * 2);
  bf16_t* znb     = (bf16_t*)alloc((size_t)NROWS * DIM * 2);
  float*  enf     = (float*)alloc((size_t)KCB * DIM * 4);
  float*  simb    = (float*)alloc((size_t)NROWS * KCBP * 4);
  float*  commit  = (float*)alloc(256);
  int*    counts  = (int*)alloc(512);

  float* out_f  = (float*)d_out;
  float* loss_p = out_f + (size_t)BATCH * 3 * 100 * 140;
  int*   idx_p  = (int*)(loss_p + 1);

  auto convT = [&](const float* W, bf16_t* WT, int K, int N, int Kpad) {
    int total = N * Kpad;
    convert_wT<<<(total + 255) / 256, 256, 0, stream>>>(W, WT, K, N, Kpad);
  };

  // ---- weight conversion (every launch; deterministic) ----
  convT(patch_w, patchT, 1200, 512, PPKP);
  convT(pred_w, predT, 512, 1200, 512);
  for (int gi = 0; gi < 12; ++gi) {
    BlkP bp = getBlk(gi < 6 ? 4 : 18, gi % 6);
    convT(bp.qkvw, qkvT[gi], 512, 1536, 512);
    convT(bp.projw, projT[gi], 512, 512, 512);
    convT(bp.w1, w1T[gi], 512, 2048, 512);
    convT(bp.w2, w2T[gi], 2048, 512, 2048);
  }
  en_kernel<<<KCBP, 256, 0, stream>>>(embed, enf, enbf);

  auto gemm = [&](int epi, const bf16_t* A, const bf16_t* BT, float* C, bf16_t* Cb,
                  const float* bias, const float* extra, int N, int K, int ldc) {
    dim3 g((N + 127) / 128, NROWS / 128);
    switch (epi) {
      case 0: gemm_bf16<0, true ><<<g, 128, 0, stream>>>(A, BT, C, Cb, bias, extra, N, K, ldc); break;
      case 1: gemm_bf16<1, true ><<<g, 128, 0, stream>>>(A, BT, C, Cb, bias, extra, N, K, ldc); break;
      case 2: gemm_bf16<2, true ><<<g, 128, 0, stream>>>(A, BT, C, Cb, bias, extra, N, K, ldc); break;
      case 3: gemm_bf16<3, true ><<<g, 128, 0, stream>>>(A, BT, C, Cb, bias, extra, N, K, ldc); break;
      case 4: gemm_bf16<4, false><<<g, 128, 0, stream>>>(A, BT, C, Cb, bias, extra, N, K, ldc); break;
      case 5: gemm_bf16<5, false><<<g, 128, 0, stream>>>(A, BT, C, Cb, bias, extra, N, K, ldc); break;
    }
  };

  // ---- patch embed: x -> patches -> xst = patches@W + b + pos ----
  {
    int total = NROWS * PPKP;
    patch_kernel<<<(total + 255) / 256, 256, 0, stream>>>(x_in, patches);
  }
  gemm(3, patches, patchT, xst, nullptr, patch_b, pos, 512, PPKP, 512);

  auto run_block = [&](int gi) {
    BlkP bp = getBlk(gi < 6 ? 4 : 18, gi % 6);
    ln_kernel<1><<<NROWS, 256, 0, stream>>>(xst, bp.ln1g, bp.ln1b, nullptr, hbuf);
    gemm(0, hbuf, qkvT[gi], qkvbuf, nullptr, bp.qkvb, nullptr, 1536, 512, 1536);
    {
      int total = BATCH * NHEAD * NTOK * HDIM;
      rope_kernel<<<(total + 255) / 256, 256, 0, stream>>>(qkvbuf, qhb, khb, vhb);
    }
    attn_kernel<<<BATCH * NHEAD, 256, 0, stream>>>(qhb, khb, vhb, Obuf);
    gemm(2, Obuf, projT[gi], xst, nullptr, bp.projb, xst, 512, 512, 512);
    ln_kernel<1><<<NROWS, 256, 0, stream>>>(xst, bp.ln2g, bp.ln2b, nullptr, hbuf);
    gemm(1, hbuf, w1T[gi], nullptr, ff1buf, bp.b1, nullptr, 2048, 512, 2048);
    gemm(2, ff1buf, w2T[gi], xst, nullptr, bp.b2, xst, 512, 2048, 512);
  };

  for (int gi = 0; gi < 6; ++gi) run_block(gi);   // encoder

  // ---- VQ ----
  ln_kernel<0><<<NROWS, 256, 0, stream>>>(xst, enc_g, enc_b, zbuf, nullptr);
  zn_kernel<<<NROWS, 256, 0, stream>>>(zbuf, znb);
  gemm(4, znb, enbf, simb, nullptr, nullptr, nullptr, KCBP, 512, KCBP);
  init_stats<<<1, 128, 0, stream>>>(commit, counts);
  vq_kernel<<<NROWS, 128, 0, stream>>>(simb, enf, zbuf, pos, xst, idx_p, commit, counts);

  for (int gi = 6; gi < 12; ++gi) run_block(gi);  // decoder

  // ---- pixel head (fused tanh + un-patch scatter) ----
  ln_kernel<1><<<NROWS, 256, 0, stream>>>(xst, dec_g, dec_b, nullptr, hbuf);
  gemm(5, hbuf, predT, out_f, nullptr, pred_b, nullptr, 1200, 512, 1200);
  finalize_kernel<<<1, 1, 0, stream>>>(commit, counts, loss_p);
}
